// RGATLayer_73778948210783
// MI455X (gfx1250) — compile-verified, matrix operand order checked by reading, not run on hardware
//
#include <hip/hip_runtime.h>
#include <hip/hip_bf16.h>
#include <math.h>

typedef __attribute__((ext_vector_type(16))) __bf16 v16bf;
typedef __attribute__((ext_vector_type(8)))  float  v8f;

#define IN_DIM  64
#define OUT_DIM 64
#define NEG_SLOPE 0.2f

__device__ __forceinline__ float lrelu(float v) {
    return v > 0.0f ? v : NEG_SLOPE * v;
}

// Portable float atomic-max via sign-split integer atomics (works for mixed signs).
__device__ __forceinline__ void atomicMaxF(float* addr, float val) {
    if (val >= 0.0f) atomicMax((int*)addr, __float_as_int(val));
    else             atomicMin((unsigned int*)addr, __float_as_uint(val));
}

// -----------------------------------------------------------------------------
// K1: h = x @ W (bf16 WMMA, f32 accum), fused a_src = h.att_src, a_dst = h.att_dst
// block = 256 threads = 8 waves; wave w handles rows [blk*128 + w*16, +16)
//
// W is pre-swizzled into LDS in *fragment-major* order: fragment f = kc*4+nt,
// lane l owns 16 contiguous bf16 (32 B) -> B fragment load = 1 aligned 32B read.
// A fragment per lane = two contiguous 8-float runs -> 4 x float4 global loads.
// -----------------------------------------------------------------------------
__global__ __launch_bounds__(256)
void gat_gemm_kernel(const float* __restrict__ x, const float* __restrict__ W,
                     const float* __restrict__ att_src, const float* __restrict__ att_dst,
                     float* __restrict__ h, float* __restrict__ asrc, float* __restrict__ adst,
                     int N)
{
    __shared__ v16bf WbSw[8 * 32];            // 8 fragments x 32 lanes x 16 bf16 = 8 KB

    // ---- cooperative swizzle fill: thread t handles (frag = t>>5, lane = t&31) ----
    {
        const int f  = threadIdx.x >> 5;      // kc*4 + nt
        const int l  = threadIdx.x & 31;
        const int kc = f >> 2;
        const int nt = f & 3;
        const int hf = l >> 4;
        const int nn = l & 15;
        v16bf frag;
        #pragma unroll
        for (int i = 0; i < 16; ++i) {
            const int k = kc * 32 + ((i < 8) ? (8 * hf + i) : (16 + 8 * hf + (i - 8)));
            frag[i] = (__bf16)W[k * OUT_DIM + nt * 16 + nn];
        }
        WbSw[f * 32 + l] = frag;
    }
    __syncthreads();

    const int wave = threadIdx.x >> 5;
    const int lane = threadIdx.x & 31;
    const int row0 = blockIdx.x * 128 + wave * 16;
    if (row0 >= N) return;                    // wave-uniform

    const int n    = lane & 15;               // column within 16-wide tile (B/C/D)
    const int half = lane >> 4;

    v8f c0 = {}, c1 = {}, c2 = {}, c3 = {};

    #pragma unroll
    for (int kc = 0; kc < 2; ++kc) {          // K = 64 in two 32-chunks
        // ---- A fragment: 16x32 bf16, row m = lane&15; two contiguous 8-float runs ----
        int arow = row0 + n;  if (arow >= N) arow = N - 1;   // clamp, keep EXEC full
        const float* xr = x + (size_t)arow * IN_DIM + kc * 32 + 8 * half;
        const float4 f0 = *(const float4*)(xr + 0);
        const float4 f1 = *(const float4*)(xr + 4);
        const float4 f2 = *(const float4*)(xr + 16);
        const float4 f3 = *(const float4*)(xr + 20);
        v16bf a;
        a[0]  = (__bf16)f0.x;  a[1]  = (__bf16)f0.y;  a[2]  = (__bf16)f0.z;  a[3]  = (__bf16)f0.w;
        a[4]  = (__bf16)f1.x;  a[5]  = (__bf16)f1.y;  a[6]  = (__bf16)f1.z;  a[7]  = (__bf16)f1.w;
        a[8]  = (__bf16)f2.x;  a[9]  = (__bf16)f2.y;  a[10] = (__bf16)f2.z;  a[11] = (__bf16)f2.w;
        a[12] = (__bf16)f3.x;  a[13] = (__bf16)f3.y;  a[14] = (__bf16)f3.z;  a[15] = (__bf16)f3.w;

        // ---- B fragments: one 32-byte LDS vector load each ----
        const v16bf b0 = WbSw[(kc * 4 + 0) * 32 + lane];
        const v16bf b1 = WbSw[(kc * 4 + 1) * 32 + lane];
        const v16bf b2 = WbSw[(kc * 4 + 2) * 32 + lane];
        const v16bf b3 = WbSw[(kc * 4 + 3) * 32 + lane];

        c0 = __builtin_amdgcn_wmma_f32_16x16x32_bf16(false, a, false, b0, (short)0, c0, false, false);
        c1 = __builtin_amdgcn_wmma_f32_16x16x32_bf16(false, a, false, b1, (short)0, c1, false, false);
        c2 = __builtin_amdgcn_wmma_f32_16x16x32_bf16(false, a, false, b2, (short)0, c2, false, false);
        c3 = __builtin_amdgcn_wmma_f32_16x16x32_bf16(false, a, false, b3, (short)0, c3, false, false);
    }

    // attention vectors, 4 columns per lane
    float attS[4], attD[4];
    #pragma unroll
    for (int nt = 0; nt < 4; ++nt) {
        attS[nt] = att_src[nt * 16 + n];
        attD[nt] = att_dst[nt * 16 + n];
    }

    // Epilogue: store h, reduce attention dots across the 16-lane half.
    #pragma unroll
    for (int r = 0; r < 8; ++r) {
        const int row = row0 + r + 8 * half;  // C/D layout: VGPR r holds M = r + 8*half
        float vals[4] = { c0[r], c1[r], c2[r], c3[r] };
        float ps = vals[0] * attS[0] + vals[1] * attS[1] + vals[2] * attS[2] + vals[3] * attS[3];
        float pd = vals[0] * attD[0] + vals[1] * attD[1] + vals[2] * attD[2] + vals[3] * attD[3];
        #pragma unroll
        for (int m = 1; m < 16; m <<= 1) {    // butterfly within the half
            ps += __shfl_xor(ps, m, 32);
            pd += __shfl_xor(pd, m, 32);
        }
        if (row < N) {
            float* hr = h + (size_t)row * OUT_DIM;
            #pragma unroll
            for (int nt = 0; nt < 4; ++nt) hr[nt * 16 + n] = vals[nt];
            if (n == 0) { asrc[row] = ps; adst[row] = pd; }
        }
    }
}

// -----------------------------------------------------------------------------
// K2: per-node init: segmax seeded with self-loop score, denom = 0
// -----------------------------------------------------------------------------
__global__ void gat_node_init_kernel(const float* __restrict__ asrc, const float* __restrict__ adst,
                                     float* __restrict__ segmax, float* __restrict__ denom, int N)
{
    int i = blockIdx.x * blockDim.x + threadIdx.x;
    if (i >= N) return;
    segmax[i] = lrelu(asrc[i] + adst[i]);
    denom[i]  = 0.0f;
}

// -----------------------------------------------------------------------------
// K3: per-edge score + atomic segment max over dst
// -----------------------------------------------------------------------------
__global__ void gat_edge_max_kernel(const int* __restrict__ src, const int* __restrict__ dst,
                                    const float* __restrict__ asrc, const float* __restrict__ adst,
                                    float* __restrict__ ebuf, float* __restrict__ segmax, int E)
{
    int e = blockIdx.x * blockDim.x + threadIdx.x;
    if (e >= E) return;
    const int s = src[e], d = dst[e];
    const float v = lrelu(asrc[s] + adst[d]);
    ebuf[e] = v;
    atomicMaxF(&segmax[d], v);
}

// -----------------------------------------------------------------------------
// K4: per-edge softmax numerator + weighted message scatter (16 lanes / edge)
// -----------------------------------------------------------------------------
__global__ __launch_bounds__(256)
void gat_edge_agg_kernel(const int* __restrict__ src, const int* __restrict__ dst,
                         const float* __restrict__ ebuf, const float* __restrict__ segmax,
                         const float* __restrict__ h,
                         float* __restrict__ denom, float* __restrict__ out, int E)
{
    int t    = blockIdx.x * blockDim.x + threadIdx.x;
    int e    = t >> 4;
    int part = t & 15;                       // 16 lanes x float4 = 64 cols
    if (e >= E) return;
    const int s = src[e], d = dst[e];
    const float p = __expf(ebuf[e] - segmax[d]);
    if (part == 0) atomicAdd(&denom[d], p);
    const float4 hv = ((const float4*)(h + (size_t)s * OUT_DIM))[part];
    float* od = out + (size_t)d * OUT_DIM + part * 4;
    atomicAdd(od + 0, p * hv.x);
    atomicAdd(od + 1, p * hv.y);
    atomicAdd(od + 2, p * hv.z);
    atomicAdd(od + 3, p * hv.w);
}

// -----------------------------------------------------------------------------
// K5: add self-loop contribution, normalize, add bias
// -----------------------------------------------------------------------------
__global__ void gat_finalize_kernel(const float* __restrict__ asrc, const float* __restrict__ adst,
                                    const float* __restrict__ segmax, const float* __restrict__ denom,
                                    const float* __restrict__ h, const float* __restrict__ bias,
                                    float* __restrict__ out, int total)
{
    int i = blockIdx.x * blockDim.x + threadIdx.x;
    if (i >= total) return;
    const int nidx = i >> 6;
    const int c    = i & 63;
    const float es = lrelu(asrc[nidx] + adst[nidx]);         // self-loop score
    const float ps = __expf(es - segmax[nidx]);
    const float den = denom[nidx] + ps + 1e-16f;
    out[i] = (out[i] + ps * h[i]) / den + bias[c];
}

// -----------------------------------------------------------------------------
extern "C" void kernel_launch(void* const* d_in, const int* in_sizes, int n_in,
                              void* d_out, int out_size, void* d_ws, size_t ws_size,
                              hipStream_t stream)
{
    const float* x        = (const float*)d_in[0];
    const int*   ei       = (const int*)  d_in[1];   // [2, E] row-major
    const float* W        = (const float*)d_in[2];
    const float* att_src  = (const float*)d_in[3];
    const float* att_dst  = (const float*)d_in[4];
    const float* bias     = (const float*)d_in[5];
    float*       out      = (float*)d_out;

    const int N = in_sizes[0] / IN_DIM;
    const int E = in_sizes[1] / 2;
    const int* src = ei;
    const int* dst = ei + E;

    // workspace layout (floats): h[N*64] | asrc[N] | adst[N] | segmax[N] | denom[N] | ebuf[E]
    float* h      = (float*)d_ws;
    float* asrc   = h + (size_t)N * OUT_DIM;
    float* adst   = asrc + N;
    float* segmax = adst + N;
    float* denom  = segmax + N;
    float* ebuf   = denom + N;

    // K1: WMMA GEMM + attention scores
    gat_gemm_kernel<<<(N + 127) / 128, 256, 0, stream>>>(x, W, att_src, att_dst,
                                                         h, asrc, adst, N);
    // zero the output accumulator (graph-capture safe)
    hipMemsetAsync(out, 0, (size_t)N * OUT_DIM * sizeof(float), stream);

    // K2: per-node init
    gat_node_init_kernel<<<(N + 255) / 256, 256, 0, stream>>>(asrc, adst, segmax, denom, N);

    // K3: per-edge scores + segment max
    gat_edge_max_kernel<<<(E + 255) / 256, 256, 0, stream>>>(src, dst, asrc, adst,
                                                             ebuf, segmax, E);

    // K4: per-edge weighted aggregation (16 lanes per edge)
    {
        long long threads = (long long)E * 16;
        int blocks = (int)((threads + 255) / 256);
        gat_edge_agg_kernel<<<blocks, 256, 0, stream>>>(src, dst, ebuf, segmax, h,
                                                        denom, out, E);
    }

    // K5: self-loop + normalize + bias
    {
        int total = N * OUT_DIM;
        gat_finalize_kernel<<<(total + 255) / 256, 256, 0, stream>>>(asrc, adst, segmax,
                                                                     denom, h, bias, out, total);
    }
}
